// GaussianRasterizer_74629351735331
// MI455X (gfx1250) — matrix operand order repaired and checked by reading, not compiled, not run on hardware
//
#include <hip/hip_runtime.h>
#include <hip/hip_bf16.h>
#include <stdint.h>

typedef __attribute__((ext_vector_type(16))) _Float16 v16h;
typedef __attribute__((ext_vector_type(8)))  float    v8f;

#define IMG_H 128
#define IMG_W 128
#define TANX 0.5f
#define TANY 0.5f
#define T_EPS 1e-4f
#define ALPHA_MIN (1.0f / 255.0f)

// Probe for the gfx1250 async global->LDS path (ASYNCcnt-tracked, no VGPR
// round-trip). Falls back to synchronous staging if the builtin is absent.
// The builtin takes generic `int*` pointers; clang's HIP mode implicitly
// converts them to the device/shared address spaces.
#if defined(__has_builtin)
#if __has_builtin(__builtin_amdgcn_global_load_async_to_lds_b32)
#define HAVE_ASYNC_LDS 1
#endif
#endif

#if defined(HAVE_ASYNC_LDS)
#if __has_builtin(__builtin_amdgcn_s_wait_asynccnt)
#define WAIT_ASYNC(k) __builtin_amdgcn_s_wait_asynccnt(k)
#else
#define WAIT_ASYNC(k) asm volatile("s_wait_asynccnt %0" :: "i"(k) : "memory")
#endif
#endif

// ---------------------------------------------------------------------------
// Kernel 1: per-Gaussian preprocessing.
// Produces unsorted SoA arrays in workspace:
//   u[0]=px u[1]=py u[2]=conA u[3]=conB u[4]=conC u[5]=op(valid-folded)
//   u[6..8]=rgb u[9]=tz     and radii (int32) directly into d_out.
// ---------------------------------------------------------------------------
__global__ void gs_preprocess(const float* __restrict__ means3D,
                              const float* __restrict__ opac,
                              const float* __restrict__ colors,
                              const float* __restrict__ scales,
                              const float* __restrict__ rots,
                              const float* __restrict__ V,   // viewmatrix 4x4 row-major
                              const float* __restrict__ P,   // projmatrix 4x4 row-major
                              float* __restrict__ u,
                              int* __restrict__ radii_out,
                              int n)
{
    int i = blockIdx.x * blockDim.x + threadIdx.x;
    if (i >= n) return;

    float mx = means3D[3 * i + 0];
    float my = means3D[3 * i + 1];
    float mz = means3D[3 * i + 2];

    // p_view = [m,1] @ V   (row-vector convention)
    float pv0 = mx * V[0] + my * V[4] + mz * V[8]  + V[12];
    float pv1 = mx * V[1] + my * V[5] + mz * V[9]  + V[13];
    float pv2 = mx * V[2] + my * V[6] + mz * V[10] + V[14];
    float tz  = pv2;

    // p_hom = [m,1] @ P
    float ph0 = mx * P[0] + my * P[4] + mz * P[8]  + P[12];
    float ph1 = mx * P[1] + my * P[5] + mz * P[9]  + P[13];
    float ph3 = mx * P[3] + my * P[7] + mz * P[11] + P[15];
    float inv = 1.0f / (ph3 + 1e-7f);
    float ppx = ph0 * inv;
    float ppy = ph1 * inv;

    // normalized quaternion -> rotation matrix
    float qr = rots[4 * i + 0], qx = rots[4 * i + 1];
    float qy = rots[4 * i + 2], qz = rots[4 * i + 3];
    float qn = 1.0f / sqrtf(qr * qr + qx * qx + qy * qy + qz * qz);
    qr *= qn; qx *= qn; qy *= qn; qz *= qn;

    float R[3][3];
    R[0][0] = 1.0f - 2.0f * (qy * qy + qz * qz);
    R[0][1] = 2.0f * (qx * qy - qr * qz);
    R[0][2] = 2.0f * (qx * qz + qr * qy);
    R[1][0] = 2.0f * (qx * qy + qr * qz);
    R[1][1] = 1.0f - 2.0f * (qx * qx + qz * qz);
    R[1][2] = 2.0f * (qy * qz - qr * qx);
    R[2][0] = 2.0f * (qx * qz - qr * qy);
    R[2][1] = 2.0f * (qy * qz + qr * qx);
    R[2][2] = 1.0f - 2.0f * (qx * qx + qy * qy);

    float sc[3] = { scales[3 * i], scales[3 * i + 1], scales[3 * i + 2] };
    float M[3][3];
    #pragma unroll
    for (int a = 0; a < 3; ++a)
        #pragma unroll
        for (int b = 0; b < 3; ++b)
            M[a][b] = R[a][b] * sc[b];

    float Sig[3][3];
    #pragma unroll
    for (int a = 0; a < 3; ++a)
        #pragma unroll
        for (int b = 0; b < 3; ++b)
            Sig[a][b] = M[a][0] * M[b][0] + M[a][1] * M[b][1] + M[a][2] * M[b][2];

    // cov_cam = W^T Sigma W, W = V[:3,:3]
    float A[3][3];
    #pragma unroll
    for (int a = 0; a < 3; ++a)
        #pragma unroll
        for (int k = 0; k < 3; ++k)
            A[a][k] = V[0 * 4 + a] * Sig[0][k] + V[1 * 4 + a] * Sig[1][k] + V[2 * 4 + a] * Sig[2][k];
    float CC[3][3];
    #pragma unroll
    for (int a = 0; a < 3; ++a)
        #pragma unroll
        for (int l = 0; l < 3; ++l)
            CC[a][l] = A[a][0] * V[0 * 4 + l] + A[a][1] * V[1 * 4 + l] + A[a][2] * V[2 * 4 + l];

    const float fx = IMG_W / (2.0f * TANX);
    const float fy = IMG_H / (2.0f * TANY);
    float txv = fminf(fmaxf(pv0 / tz, -1.3f * TANX), 1.3f * TANX) * tz;
    float tyv = fminf(fmaxf(pv1 / tz, -1.3f * TANY), 1.3f * TANY) * tz;
    float itz = 1.0f / tz;
    float J0[3] = { fx * itz, 0.0f, -fx * txv * itz * itz };
    float J1[3] = { 0.0f, fy * itz, -fy * tyv * itz * itz };

    float t0[3], t1[3];
    #pragma unroll
    for (int k = 0; k < 3; ++k) {
        t0[k] = J0[0] * CC[0][k] + J0[1] * CC[1][k] + J0[2] * CC[2][k];
        t1[k] = J1[0] * CC[0][k] + J1[1] * CC[1][k] + J1[2] * CC[2][k];
    }
    float c00 = t0[0] * J0[0] + t0[1] * J0[1] + t0[2] * J0[2];
    float c01 = t0[0] * J1[0] + t0[1] * J1[1] + t0[2] * J1[2];
    float c11 = t1[0] * J1[0] + t1[1] * J1[1] + t1[2] * J1[2];

    float aa = c00 + 0.3f;
    float cc = c11 + 0.3f;
    float bb = c01;
    float det = aa * cc - bb * bb;
    bool valid = (tz > 0.2f) && (det > 0.0f);
    float det_s = (det > 0.0f) ? det : 1.0f;
    float conA = cc / det_s, conB = -bb / det_s, conC = aa / det_s;
    float mid = 0.5f * (aa + cc);
    float lam = mid + sqrtf(fmaxf(0.1f, mid * mid - det));
    int rad = valid ? (int)ceilf(3.0f * sqrtf(lam)) : 0;

    float px = ((ppx + 1.0f) * IMG_W - 1.0f) * 0.5f;
    float py = ((ppy + 1.0f) * IMG_H - 1.0f) * 0.5f;

    u[0 * n + i] = px;
    u[1 * n + i] = py;
    u[2 * n + i] = conA;
    u[3 * n + i] = conB;
    u[4 * n + i] = conC;
    u[5 * n + i] = valid ? opac[i] : 0.0f;   // invalid folded into zero opacity
    u[6 * n + i] = colors[3 * i + 0];
    u[7 * n + i] = colors[3 * i + 1];
    u[8 * n + i] = colors[3 * i + 2];
    u[9 * n + i] = tz;
    radii_out[i] = rad;
}

// ---------------------------------------------------------------------------
// Kernel 2: stable ascending rank sort by tz (O(N^2), N=2048 -> 4M cmp) and
// scatter all 10 SoA arrays into sorted order.
// ---------------------------------------------------------------------------
__global__ void gs_sort(const float* __restrict__ u, float* __restrict__ s, int n)
{
    __shared__ float stz[256];
    int i = blockIdx.x * 256 + threadIdx.x;
    float my_tz = u[9 * n + i];
    int rank = 0;
    for (int t = 0; t < n; t += 256) {
        stz[threadIdx.x] = u[9 * n + t + threadIdx.x];
        __syncthreads();
        for (int j = 0; j < 256; ++j) {
            float o = stz[j];
            int jj = t + j;
            rank += ((o < my_tz) || (o == my_tz && jj < i)) ? 1 : 0;
        }
        __syncthreads();
    }
    #pragma unroll
    for (int f = 0; f < 10; ++f)
        s[f * n + rank] = u[f * n + i];
}

// ---------------------------------------------------------------------------
// Kernel 3: WMMA rasterizer. One wave = 16 consecutive pixels (rows M of the
// 16x16 accumulator); every 32 sorted Gaussians form one K-chunk of a
// V_WMMA_F32_16X16X32_F16. B columns 0..3 = [r,g,b,tz], rest zero.
// Tiles of 256 Gaussians are double-buffered in LDS via async global->LDS
// copies (when available) so the next tile's fill overlaps this tile's
// alpha/transmittance chain + WMMA.
// ---------------------------------------------------------------------------
__global__ void __launch_bounds__(256)
gs_rasterize(const float* __restrict__ s, const float* __restrict__ bg,
             float* __restrict__ out, int n)
{
    __shared__ float sh[2][10 * 256];   // double-buffered tile, 20 KB

    const int tid   = threadIdx.x;
    const int lane  = tid & 31;
    const int wave  = tid >> 5;
    const int wbase = blockIdx.x * 128 + wave * 16;   // first pixel of this wave
    const int pixel = wbase + (lane & 15);
    const float gx = (float)(pixel & (IMG_W - 1));
    const float gy = (float)(pixel >> 7);

    const int hiHalf = (lane & 16) ? 1 : 0;   // lanes 16..31 own the upper K-subset
    const int nfeat  = lane & 15;             // accumulator column owned by this lane

    v8f acc = {};
    float T = 1.0f;
    int cnt = 0;

    const int TILE = 256;
    int cur = 0;

#if defined(HAVE_ASYNC_LDS)
    // Kick off the first tile's async fill.
    #pragma unroll
    for (int f = 0; f < 10; ++f)
        __builtin_amdgcn_global_load_async_to_lds_b32(
            (int*)(uintptr_t)(s + (size_t)f * n + tid),
            (int*)&sh[0][f * 256 + tid], 0, 0);
#endif

    for (int t = 0; t < n; t += TILE) {
#if defined(HAVE_ASYNC_LDS)
        if (t + TILE < n) {
            // Issue next tile into the other buffer, then wait only for the
            // 10 outstanding copies of the current tile (in-order completion).
            #pragma unroll
            for (int f = 0; f < 10; ++f)
                __builtin_amdgcn_global_load_async_to_lds_b32(
                    (int*)(uintptr_t)(s + (size_t)f * n + (t + TILE) + tid),
                    (int*)&sh[cur ^ 1][f * 256 + tid], 0, 0);
            WAIT_ASYNC(10);
        } else {
            WAIT_ASYNC(0);
        }
        __syncthreads();
#else
        #pragma unroll
        for (int f = 0; f < 10; ++f)
            sh[cur][f * 256 + tid] = s[f * n + t + tid];
        __syncthreads();
#endif
        const float* shc = &sh[cur][0];

        for (int c = 0; c < TILE; c += 32) {
            // Sequential transmittance chain over the 32-Gaussian chunk.
            float w32[32];
            #pragma unroll
            for (int k = 0; k < 32; ++k) {
                int idx = c + k;
                float dx = shc[0 * 256 + idx] - gx;
                float dy = shc[1 * 256 + idx] - gy;
                float power = -0.5f * (shc[2 * 256 + idx] * dx * dx +
                                       shc[4 * 256 + idx] * dy * dy)
                              - shc[3 * 256 + idx] * dx * dy;
                float alpha = fminf(0.99f, shc[5 * 256 + idx] * __expf(power));
                alpha = (power <= 0.0f && alpha >= ALPHA_MIN) ? alpha : 0.0f;
                float w = (T > T_EPS) ? alpha * T : 0.0f;
                T *= (1.0f - alpha);
                cnt += (w > 0.0f) ? 1 : 0;
                w32[k] = w;
            }

            // Pack A fragment (16x32 f16): lanes 0-15 K={0..7,16..23},
            // lanes 16-31 K={8..15,24..31}; indices are compile-time constant,
            // half-wave selection via cndmask (no scratch).
            v16h a;
            #pragma unroll
            for (int e = 0; e < 16; ++e) {
                const int klo = (e & 7) + ((e & 8) << 1);
                float wlo = w32[klo];
                float whi = w32[klo + 8];
                a[e] = (_Float16)(hiHalf ? whi : wlo);
            }

            // Pack B fragment (32x16 f16): lane column N carries feature N
            // (0..2 rgb, 3 tz, others zero). lanes 0-15 K={0..7,16..23}, 16-31 K={8..15,24..31}.
            v16h b;
            const int kb = (lane & 16) >> 1;
            #pragma unroll
            for (int e = 0; e < 16; ++e) {
                float fv = 0.0f;
                if (nfeat < 4)
                    fv = shc[(6 + nfeat) * 256 + (c + kb + (e & 7) + ((e & 8) << 1))];
                b[e] = (_Float16)fv;
            }

            acc = __builtin_amdgcn_wmma_f32_16x16x32_f16(
                /*neg_a=*/false, a, /*neg_b=*/false, b,
                /*c_mod=*/(short)0, acc, /*reuse_a=*/false, /*reuse_b=*/false);
        }
        __syncthreads();
        cur ^= 1;
    }

    // Write-out. Accumulator: VGPR r holds row M = r + 8*hiHalf, column nfeat.
    const int HW = IMG_H * IMG_W;
    float bgv = (nfeat < 3) ? bg[nfeat] : 0.0f;
    #pragma unroll
    for (int r = 0; r < 8; ++r) {
        int M = r + (hiHalf << 3);
        float Tf = __shfl(T, M, 32);     // final transmittance of pixel wbase+M
        float v  = acc[r];
        if (nfeat < 3)
            out[nfeat * HW + wbase + M] = v + Tf * bgv;   // color (3,H,W)
        else if (nfeat == 3)
            out[3 * HW + wbase + M] = v;                  // depth (1,H,W)
    }
    if (lane < 16)
        ((int*)out)[4 * HW + n + (wbase + lane)] = cnt;   // counter (H,W) int32
}

// ---------------------------------------------------------------------------
extern "C" void kernel_launch(void* const* d_in, const int* in_sizes, int n_in,
                              void* d_out, int out_size, void* d_ws, size_t ws_size,
                              hipStream_t stream)
{
    const float* bg      = (const float*)d_in[0];
    const float* means3D = (const float*)d_in[1];
    // d_in[2] = means2D (unused by the reference math)
    const float* opac    = (const float*)d_in[3];
    const float* colors  = (const float*)d_in[4];
    const float* scales  = (const float*)d_in[5];
    const float* rots    = (const float*)d_in[6];
    const float* viewm   = (const float*)d_in[7];
    const float* projm   = (const float*)d_in[8];

    const int n = in_sizes[1] / 3;          // number of Gaussians (2048)
    const int HW = IMG_H * IMG_W;

    float* u = (float*)d_ws;                // unsorted SoA: 10*n floats
    float* s = u + 10 * n;                  // sorted SoA:   10*n floats

    float* outf = (float*)d_out;            // color(3HW) | depth(HW) | radii(n) | counter(HW)
    int* radii_out = (int*)d_out + 4 * HW;

    gs_preprocess<<<(n + 255) / 256, 256, 0, stream>>>(
        means3D, opac, colors, scales, rots, viewm, projm, u, radii_out, n);

    gs_sort<<<n / 256, 256, 0, stream>>>(u, s, n);

    gs_rasterize<<<HW / 128, 256, 0, stream>>>(s, bg, outf, n);
}
